// FlashAttentionModule_53377853554835
// MI455X (gfx1250) — matrix operand order repaired and checked by reading, not compiled, not run on hardware
//
#include <hip/hip_runtime.h>
#include <hip/hip_bf16.h>

// ---------------------------------------------------------------------------
// MI455X (gfx1250) attention block: Q/K/V proj -> RoPE -> causal flash attn
// -> output proj.  ~0.7 TFLOP of matrix math vs ~25us of HBM traffic at
// 23.3 TB/s => compute bound => everything runs on v_wmma_f32_16x16x32_bf16
// (f32 accumulate).  f32 operands are pre-converted to bf16 once (memory
// bound, negligible), so the GEMM hot loop is pure: async global->LDS b128
// staging (double buffered, ASYNCcnt-throttled via s_wait_asynccnt) feeding
// 8 WMMAs per wave per 32-deep k-step.
// ---------------------------------------------------------------------------

typedef __attribute__((ext_vector_type(4)))  int    vi4;
typedef __attribute__((ext_vector_type(16))) __bf16 v16bf;
typedef __attribute__((ext_vector_type(2)))  __bf16 v2bf;
typedef __attribute__((ext_vector_type(8)))  float  v8f;

union Frag { vi4 i[2]; v16bf v; };

#define BQ   2
#define TQ   2048
#define HQ   32
#define DHQ  128
#define CQ   4096

// ---------------- CDNA5 async global->LDS staging (inline asm) ----------------
// The clang builtins use HIP language address spaces we cannot spell from
// source, so emit the ISA ops directly (CDNA5 ISA).
typedef __attribute__((address_space(3))) unsigned short lds_u16_t;

__device__ __forceinline__ void ld128_to_lds(unsigned short* lds,
                                             const unsigned short* g) {
    // 32-bit wave-relative LDS byte address for VDST.
    unsigned loff = (unsigned)(unsigned long long)(lds_u16_t*)lds;
    asm volatile("global_load_async_to_lds_b128 %0, %1, off"
                 :: "v"(loff), "v"(g)
                 : "memory");
}
__device__ __forceinline__ void wait_async_le4() {
    asm volatile("s_wait_asynccnt 0x4" ::: "memory");
}
__device__ __forceinline__ void wait_async_0() {
    asm volatile("s_wait_asynccnt 0x0" ::: "memory");
}

// ---------------- bf16 helpers ----------------
__device__ __forceinline__ unsigned short f2bf(float f) {
    unsigned u = __float_as_uint(f);
    unsigned r = u + 0x7FFFu + ((u >> 16) & 1u);   // round-to-nearest-even
    return (unsigned short)(r >> 16);
}
__device__ __forceinline__ unsigned pack2bf(float x, float y) {
#if defined(__has_builtin) && __has_builtin(__builtin_amdgcn_cvt_pk_bf16_f32)
    v2bf p = __builtin_amdgcn_cvt_pk_bf16_f32(x, y);
    return __builtin_bit_cast(unsigned, p);
#else
    return (unsigned)f2bf(x) | ((unsigned)f2bf(y) << 16);
#endif
}
__device__ __forceinline__ v8f wmma_bf16(const Frag& a, const Frag& b, v8f c) {
    return __builtin_amdgcn_wmma_f32_16x16x32_bf16(
        false, a.v, false, b.v, (short)0, c, false, false);
}

// ---------------------------------------------------------------------------
// Elementwise f32 -> bf16 (vectorized b128 in / b64 out).  Memory bound.
// ---------------------------------------------------------------------------
__global__ __launch_bounds__(256) void cvt_bf16_kernel(
    const float* __restrict__ X, unsigned short* __restrict__ Y, int n4)
{
    int i = blockIdx.x * 256 + threadIdx.x;
    if (i >= n4) return;
    float4 v = reinterpret_cast<const float4*>(X)[i];
    reinterpret_cast<uint2*>(Y)[i] =
        make_uint2(pack2bf(v.x, v.y), pack2bf(v.z, v.w));
}

// ---------------------------------------------------------------------------
// bf16 NT GEMM: C[m,n] = sum_k A[m,k] * W[n,k], f32 accumulate.
// 128x128 tile / 256 threads (8 waves); wave = 32x64 quadrant = 2x4 WMMA
// tiles; K stepped by 32; double-buffered LDS filled by async b128 loads.
// mode 0: row-major [M,N] output.  mode 1: [B,H,T,Dh] head-layout output.
// ---------------------------------------------------------------------------
__global__ __launch_bounds__(256) void gemm_bf16_kernel(
    const unsigned short* __restrict__ A, const unsigned short* __restrict__ W,
    float* __restrict__ Cout, int M, int N, int K, int mode)
{
    __shared__ __align__(16) unsigned short As[2][128 * 40];  // [row][k]
    __shared__ __align__(16) unsigned short Bs[2][128 * 40];  // [n]  [k]

    const int tid  = threadIdx.x;
    const int lane = tid & 31;
    const int wave = tid >> 5;          // 0..7
    const int half = lane >> 4;
    const int ln   = lane & 15;
    const int wr   = (wave >> 1) * 32;  // wave row offset (0,32,64,96)
    const int wc   = (wave & 1) * 64;   // wave col offset (0,64)
    const int rowBase = blockIdx.y * 128;
    const int colBase = blockIdx.x * 128;

    // Stage one 128x32 bf16 tile of A and W: 512 b128 chunks each,
    // 2 per thread per matrix (4 async wave-instructions per wave).
    auto stage = [&](int buf, int kt) {
        const unsigned short* ap = A + (size_t)rowBase * K + (size_t)kt * 32;
        const unsigned short* wp = W + (size_t)colBase * K + (size_t)kt * 32;
        #pragma unroll
        for (int i = 0; i < 2; ++i) {
            int idx = tid + i * 256;          // 0..511
            int r = idx >> 2, c = idx & 3;    // 4 b128 per 32-wide row
            ld128_to_lds(&As[buf][r * 40 + c * 8], ap + (size_t)r * K + c * 8);
            ld128_to_lds(&Bs[buf][r * 40 + c * 8], wp + (size_t)r * K + c * 8);
        }
    };

    v8f acc[2][4] = {};
    const int kTiles = K >> 5;

    stage(0, 0);
    for (int kt = 0; kt < kTiles; ++kt) {
        const int buf = kt & 1;
        if (kt + 1 < kTiles) { stage(buf ^ 1, kt + 1); wait_async_le4(); }
        else                 { wait_async_0(); }
        __syncthreads();

        Frag af[2], bfr[4];
        #pragma unroll
        for (int i = 0; i < 2; ++i) {   // A frag: lane=M, K split per half
            int row = wr + i * 16 + ln;
            af[i].i[0] = *reinterpret_cast<const vi4*>(&As[buf][row * 40 + half * 8]);
            af[i].i[1] = *reinterpret_cast<const vi4*>(&As[buf][row * 40 + 16 + half * 8]);
        }
        #pragma unroll
        for (int j = 0; j < 4; ++j) {   // B frag: lane=N, K split per half
            int col = wc + j * 16 + ln;
            bfr[j].i[0] = *reinterpret_cast<const vi4*>(&Bs[buf][col * 40 + half * 8]);
            bfr[j].i[1] = *reinterpret_cast<const vi4*>(&Bs[buf][col * 40 + 16 + half * 8]);
        }
        #pragma unroll
        for (int i = 0; i < 2; ++i)
            #pragma unroll
            for (int j = 0; j < 4; ++j)
                acc[i][j] = wmma_bf16(af[i], bfr[j], acc[i][j]);
        __syncthreads();
    }

    // Epilogue: C/D layout M = r + 8*half, N = ln.  Mode test hoisted; one
    // base pointer per wave, immediate-offset b32 stores.
    if (mode == 0) {
        float* base = Cout + (size_t)(rowBase + wr + half * 8) * N
                           + colBase + wc + ln;
        #pragma unroll
        for (int i = 0; i < 2; ++i)
            #pragma unroll
            for (int r = 0; r < 8; ++r) {
                float* p = base + (size_t)(i * 16 + r) * N;
                #pragma unroll
                for (int j = 0; j < 4; ++j) p[j * 16] = acc[i][j][r];
            }
    } else {
        // 128-wide tile == exactly one head (Dh=128, colBase 128-aligned);
        // 128 rows stay inside one batch (T=2048, rowBase 128-aligned).
        const int b  = rowBase >> 11;
        const int t0 = rowBase & (TQ - 1);
        const int h  = colBase >> 7;
        float* base = Cout + (((size_t)(b * HQ + h)) * TQ + t0 + wr + half * 8) * DHQ
                           + wc + ln;
        #pragma unroll
        for (int i = 0; i < 2; ++i)
            #pragma unroll
            for (int r = 0; r < 8; ++r) {
                float* p = base + (size_t)(i * 16 + r) * DHQ;
                #pragma unroll
                for (int j = 0; j < 4; ++j) p[j * 16] = acc[i][j][r];
            }
    }
}

// ---------------------------------------------------------------------------
// RoPE (interleaved pairs, angle index (2j)%64) + f32 -> bf16 convert.
// do_rope==0: plain convert (for V).  Layout [B,H,T,Dh].
// ---------------------------------------------------------------------------
__global__ __launch_bounds__(256) void rope_cvt_kernel(
    const float* __restrict__ X, unsigned short* __restrict__ Y, int do_rope)
{
    size_t idx = (size_t)blockIdx.x * blockDim.x + threadIdx.x;  // pair index
    const size_t total = (size_t)BQ * HQ * TQ * (DHQ / 2);
    if (idx >= total) return;
    int j = (int)(idx & 63);
    int t = (int)((idx >> 6) & (TQ - 1));
    size_t base = (idx >> 6) * DHQ + 2 * j;
    float x1 = X[base], x2 = X[base + 1];
    if (do_rope) {
        const float logTheta = 9.9034876f;             // ln(20000)
        float e    = (float)((2 * j) & 63) * (1.0f / 64.0f);
        float ang  = (float)t * __expf(-e * logTheta); // t * theta^(-i/64)
        float c = __cosf(ang), s = __sinf(ang);
        Y[base]     = f2bf(x1 * c - x2 * s);
        Y[base + 1] = f2bf(x1 * s + x2 * c);
    } else {
        Y[base]     = f2bf(x1);
        Y[base + 1] = f2bf(x2);
    }
}

// ---------------------------------------------------------------------------
// Causal flash attention.  Workgroup = (b, h, 64 q rows); 4 waves x 16 rows.
// 32-key tiles: K staged async (row-major), V staged transposed; S = Q K^T
// (8 WMMAs/wave), online softmax via shfl_xor row reductions, P re-laid out
// through per-wave LDS into an A-fragment, O += P V (8 WMMAs/wave).
// ---------------------------------------------------------------------------
__global__ __launch_bounds__(128) void attn_kernel(
    const unsigned short* __restrict__ Q, const unsigned short* __restrict__ Kg,
    const unsigned short* __restrict__ Vg, float* __restrict__ Out)
{
    __shared__ __align__(16) unsigned short Ks[32 * 136];   // [key][dh]
    __shared__ __align__(16) unsigned short Vt[128 * 40];   // [dh][key]
    __shared__ __align__(16) unsigned short Ps[4][16 * 40]; // per-wave P tile

    const int tid  = threadIdx.x;
    const int lane = tid & 31;
    const int wave = tid >> 5;
    const int half = lane >> 4;
    const int ln   = lane & 15;
    const int qbase = blockIdx.x * 64;
    const int h = blockIdx.y, b = blockIdx.z;
    const size_t headOff = ((size_t)(b * HQ + h)) * TQ * DHQ;

    // Preload this wave's Q rows as 4 A-fragments (Dh = 4 x 32).
    Frag qf[4];
    {
        int t = qbase + wave * 16 + ln;
        const unsigned short* qp = Q + headOff + (size_t)t * DHQ;
        #pragma unroll
        for (int f = 0; f < 4; ++f) {
            qf[f].i[0] = *reinterpret_cast<const vi4*>(qp + f * 32 + half * 8);
            qf[f].i[1] = *reinterpret_cast<const vi4*>(qp + f * 32 + 16 + half * 8);
        }
    }

    float mrow[8], lrow[8];
    v8f Oacc[8] = {};
    #pragma unroll
    for (int r = 0; r < 8; ++r) { mrow[r] = -3.0e38f; lrow[r] = 0.0f; }

    const float scale = 0.088388347648318447f;   // 1/sqrt(128)
    const int nkt = (qbase >> 5) + 2;            // causal: keys <= qbase+63

    for (int kt = 0; kt < nkt; ++kt) {
        __syncthreads();
        // K tile (32x128 bf16): async b128, 4 per thread.
        const unsigned short* kp = Kg + headOff + (size_t)(kt * 32) * DHQ;
        #pragma unroll
        for (int i = 0; i < 4; ++i) {
            int idx = tid + i * 128;
            int kr = idx >> 4, c8 = idx & 15;
            ld128_to_lds(&Ks[kr * 136 + c8 * 8], kp + kr * DHQ + c8 * 8);
        }
        // V tile transposed -> Vt[dh][key] (needs a register transpose).
        const unsigned short* vp = Vg + headOff + (size_t)(kt * 32) * DHQ;
        #pragma unroll
        for (int i = 0; i < 4; ++i) {
            int idx = tid + i * 128;
            int kr = idx >> 4, c8 = idx & 15;
            vi4 vv = *reinterpret_cast<const vi4*>(vp + kr * DHQ + c8 * 8);
            const unsigned short* u = reinterpret_cast<const unsigned short*>(&vv);
            #pragma unroll
            for (int e = 0; e < 8; ++e) Vt[(c8 * 8 + e) * 40 + kr] = u[e];
        }
        wait_async_0();
        __syncthreads();

        // S = Q K^T over two 16-key column tiles.
        v8f S0 = {}, S1 = {};
        #pragma unroll
        for (int kc = 0; kc < 4; ++kc) {
            Frag b0, b1;
            b0.i[0] = *reinterpret_cast<const vi4*>(&Ks[ln * 136 + kc * 32 + half * 8]);
            b0.i[1] = *reinterpret_cast<const vi4*>(&Ks[ln * 136 + kc * 32 + 16 + half * 8]);
            b1.i[0] = *reinterpret_cast<const vi4*>(&Ks[(16 + ln) * 136 + kc * 32 + half * 8]);
            b1.i[1] = *reinterpret_cast<const vi4*>(&Ks[(16 + ln) * 136 + kc * 32 + 16 + half * 8]);
            S0 = wmma_bf16(qf[kc], b0, S0);
            S1 = wmma_bf16(qf[kc], b1, S1);
        }

        // Online softmax; row m = r + 8*half lives across a 16-lane half.
        const int tq_base = qbase + wave * 16;
        #pragma unroll
        for (int r = 0; r < 8; ++r) {
            int tq = tq_base + r + half * 8;
            int c0 = kt * 32 + ln;
            float s0 = S0[r] * scale; s0 = (c0      > tq) ? -3.0e38f : s0;
            float s1 = S1[r] * scale; s1 = (c0 + 16 > tq) ? -3.0e38f : s1;
            float mx = fmaxf(s0, s1);
            #pragma unroll
            for (int m = 1; m < 16; m <<= 1) mx = fmaxf(mx, __shfl_xor(mx, m, 32));
            float mnew  = fmaxf(mrow[r], mx);
            float alpha = __expf(mrow[r] - mnew);
            float p0 = __expf(s0 - mnew);
            float p1 = __expf(s1 - mnew);
            float rs = p0 + p1;
            #pragma unroll
            for (int m = 1; m < 16; m <<= 1) rs += __shfl_xor(rs, m, 32);
            lrow[r] = lrow[r] * alpha + rs;
            mrow[r] = mnew;
            #pragma unroll
            for (int c = 0; c < 8; ++c) Oacc[c][r] = Oacc[c][r] * alpha;
            int rr = r + half * 8;
            Ps[wave][rr * 40 + ln]      = f2bf(p0);
            Ps[wave][rr * 40 + 16 + ln] = f2bf(p1);
        }
        __syncthreads();   // orders Ps writes before the A-frag reload

        // P (16x32 keys) as A-fragment; V B-fragments from transposed tile.
        Frag pf;
        pf.i[0] = *reinterpret_cast<const vi4*>(&Ps[wave][ln * 40 + half * 8]);
        pf.i[1] = *reinterpret_cast<const vi4*>(&Ps[wave][ln * 40 + 16 + half * 8]);
        #pragma unroll
        for (int c = 0; c < 8; ++c) {
            Frag vf;
            int dh = c * 16 + ln;
            vf.i[0] = *reinterpret_cast<const vi4*>(&Vt[dh * 40 + half * 8]);
            vf.i[1] = *reinterpret_cast<const vi4*>(&Vt[dh * 40 + 16 + half * 8]);
            Oacc[c] = wmma_bf16(pf, vf, Oacc[c]);
        }
    }

    // Normalize and write [B,T,C] with heads re-interleaved.
    #pragma unroll
    for (int r = 0; r < 8; ++r) {
        float inv = 1.0f / lrow[r];
        int t = qbase + wave * 16 + r + half * 8;
        float* op = Out + ((size_t)b * TQ + t) * CQ + h * DHQ;
        #pragma unroll
        for (int c = 0; c < 8; ++c) op[c * 16 + ln] = Oacc[c][r] * inv;
    }
}

// ---------------------------------------------------------------------------
// Workspace (224 MiB): tmpf f32 (64) | qb | kb | vb | xb | wb bf16 (32 each).
// X, each W, attn all have the same element count NE = 16,777,216.
// ---------------------------------------------------------------------------
extern "C" void kernel_launch(void* const* d_in, const int* in_sizes, int n_in,
                              void* d_out, int out_size, void* d_ws, size_t ws_size,
                              hipStream_t stream)
{
    const float* X  = (const float*)d_in[0];
    const float* Wq = (const float*)d_in[1];
    const float* Wk = (const float*)d_in[2];
    const float* Wv = (const float*)d_in[3];
    const float* Wo = (const float*)d_in[4];
    float* out = (float*)d_out;

    const size_t NE = (size_t)BQ * TQ * CQ;
    float*          tmpf = (float*)d_ws;
    unsigned short* qb   = (unsigned short*)((char*)d_ws + NE * sizeof(float));
    unsigned short* kb   = qb + NE;
    unsigned short* vb   = kb + NE;
    unsigned short* xb   = vb + NE;
    unsigned short* wb   = xb + NE;

    const int n4 = (int)(NE / 4);
    dim3 bCvt(256),  gCvt((unsigned)((n4 + 255) / 256));
    dim3 bGemm(256), gGemm(CQ / 128, (BQ * TQ) / 128);
    dim3 bRope(256), gRope((unsigned)((NE / 2 + 255) / 256));
    dim3 bAttn(128), gAttn(TQ / 64, HQ, BQ);

    cvt_bf16_kernel<<<gCvt, bCvt, 0, stream>>>(X, xb, n4);

    cvt_bf16_kernel<<<gCvt, bCvt, 0, stream>>>(Wq, wb, n4);
    gemm_bf16_kernel<<<gGemm, bGemm, 0, stream>>>(xb, wb, tmpf, BQ * TQ, CQ, CQ, 1);
    rope_cvt_kernel<<<gRope, bRope, 0, stream>>>(tmpf, qb, 1);

    cvt_bf16_kernel<<<gCvt, bCvt, 0, stream>>>(Wk, wb, n4);
    gemm_bf16_kernel<<<gGemm, bGemm, 0, stream>>>(xb, wb, tmpf, BQ * TQ, CQ, CQ, 1);
    rope_cvt_kernel<<<gRope, bRope, 0, stream>>>(tmpf, kb, 1);

    cvt_bf16_kernel<<<gCvt, bCvt, 0, stream>>>(Wv, wb, n4);
    gemm_bf16_kernel<<<gGemm, bGemm, 0, stream>>>(xb, wb, tmpf, BQ * TQ, CQ, CQ, 1);
    rope_cvt_kernel<<<gRope, bRope, 0, stream>>>(tmpf, vb, 0);

    attn_kernel<<<gAttn, bAttn, 0, stream>>>(qb, kb, vb, tmpf);

    cvt_bf16_kernel<<<gCvt, bCvt, 0, stream>>>(tmpf, xb, n4);
    cvt_bf16_kernel<<<gCvt, bCvt, 0, stream>>>(Wo, wb, n4);
    gemm_bf16_kernel<<<gGemm, bGemm, 0, stream>>>(xb, wb, out, BQ * TQ, CQ, CQ, 0);
}
